// EncoderLayer_56238301774416
// MI455X (gfx1250) — matrix operand order repaired and checked by reading, not compiled, not run on hardware
//
#include <hip/hip_runtime.h>
#include <hip/hip_bf16.h>
#include <math.h>

typedef _Float16 f16;
typedef __attribute__((ext_vector_type(16))) _Float16 v16h;
typedef __attribute__((ext_vector_type(8)))  _Float16 v8h;
typedef __attribute__((ext_vector_type(8)))  float    v8f;

#define D_MODEL 512
#define D_FF    2048
#define N_EXP   992
#define BS      64
#define LEN     256
#define ROWS    (BS * LEN)

// ---------------------------------------------------------------------------
// CDNA5 async global->LDS copy support (ASYNCcnt-tracked DMA staging)
// Builtin signature (decoded from clang diagnostic): 
//   void __builtin_amdgcn_global_load_async_to_lds_b128(
//       b128i AS(1)* gsrc, b128i AS(3)* ldst, imm int offset, imm int cpol)
// where b128i = int __attribute__((vector_size(16)))
// ---------------------------------------------------------------------------
#if __has_builtin(__builtin_amdgcn_global_load_async_to_lds_b128) && \
    __has_builtin(__builtin_amdgcn_s_wait_asynccnt)
#define CDNA5_ASYNC 1
typedef int b128i __attribute__((vector_size(16)));
typedef __attribute__((address_space(1))) b128i glob_b128;
typedef __attribute__((address_space(3))) b128i lds_b128;

// Copies 16B/lane; OFF (bytes) is added to BOTH the global and LDS address.
template <int OFF>
__device__ __forceinline__ void async_cp16(const void* g, unsigned lds_off) {
  __builtin_amdgcn_global_load_async_to_lds_b128(
      (glob_b128*)(unsigned long long)(size_t)g,
      (lds_b128*)(unsigned long long)lds_off, OFF, 0);
}
#endif

// ---------------------------------------------------------------------------
// WMMA helper (CDNA5 wave32, 16x16x32 f16 -> f32)
// ---------------------------------------------------------------------------
__device__ __forceinline__ v8f wmma_f16(v16h a, v16h b, v8f c) {
  return __builtin_amdgcn_wmma_f32_16x16x32_f16(
      false, a, false, b, (short)0, c, false, false);
}

// A fragment: 16x32 f16 tile, LDS row-major (stride lds halves)
__device__ __forceinline__ v16h frag_a(const f16* __restrict__ tile, int lds, int lane) {
  const int m = lane & 15;
  const int h = (lane >> 4) & 1;
  const f16* p = tile + m * lds + 16 * h;
  v16h r;
#pragma unroll
  for (int j = 0; j < 8; ++j) {
    r[j]     = p[j];
    r[8 + j] = p[8 + j];
  }
  return r;
}

// B fragment: 32x16 tile from n-major LDS (Bs[n*lds + k]); elem e -> K = e + 16h
__device__ __forceinline__ v16h frag_b(const f16* __restrict__ Bs, int lds, int n0, int lane) {
  const int n = lane & 15;
  const int h = (lane >> 4) & 1;
  const f16* p = Bs + (n0 + n) * lds + 16 * h;
  v16h r;
#pragma unroll
  for (int e = 0; e < 16; ++e) r[e] = p[e];
  return r;
}

// ---------------------------------------------------------------------------
// Batched WMMA GEMM, tile 128x64 per block, 128 threads (4 waves x 32 rows).
// Async path: LDS ping-pong double-buffer, DMA copies for tile k+1 issued
// while tile k is consumed by WMMA; one s_wait_asynccnt + barrier per step.
// ---------------------------------------------------------------------------
template <bool TRANSB>
__global__ void __launch_bounds__(128)
gemm_wmma_kernel(const f16* __restrict__ A, const f16* __restrict__ B,
                 float* __restrict__ outF, f16* __restrict__ outH,
                 const float* __restrict__ bias,
                 const float* __restrict__ mulsrc,
                 const float* __restrict__ resid,
                 const f16* __restrict__ gtab, const int* __restrict__ gidx,
                 const int* __restrict__ agather,
                 int M, int N, int K,
                 long sA, long sB, long sC,
                 float alpha, int act)
{
  const int bz   = blockIdx.z;
  const f16* Ab  = A + (long)bz * sA;
  const f16* Bb  = B + (long)bz * sB;
  const long cb  = (long)bz * sC;
  const int m0   = blockIdx.x * 128;
  const int n0   = blockIdx.y * 64;
  const int tid  = threadIdx.x;
  const int lane = tid & 31;
  const int wave = tid >> 5;

  // per-thread A source row (fixed across K loop): clamp OOB rows to 0
  const int gmA = m0 + tid;
  long arow = 0;
  if (gmA < M) {
    arow = agather ? (long)agather[(long)bz * M + gmA] : (long)gmA;
  }
  const f16* aptr = Ab + arow * (long)K;

  // per-thread B source pointer (fixed across K loop)
  const f16* bptr;
  int bk2 = 0, bc8 = 0, bnoff = 0;
  if (TRANSB) {
    const int bn = tid >> 1;
    const int bk = (tid & 1) * 16;
    bptr  = Bb + (long)(n0 + bn) * K + bk;
    bnoff = bn * 40 + bk;
  } else {
    bk2 = (tid >> 3) * 2;
    bc8 = (tid & 7) * 8;
    bptr = Bb + (long)bk2 * N + (n0 + bc8);
  }

  v8f acc[2][4];
#pragma unroll
  for (int i = 0; i < 2; ++i)
#pragma unroll
    for (int j = 0; j < 4; ++j)
#pragma unroll
      for (int r = 0; r < 8; ++r) acc[i][j][r] = 0.0f;

#ifdef CDNA5_ASYNC
  // ================= async double-buffered pipeline =================
  __shared__ __align__(16) f16 As[2][128 * 40];
  __shared__ __align__(16) f16 Bs[2][64 * 40];
  constexpr int NISS = TRANSB ? 6 : 4;     // async copies issued per step

  const unsigned aoff0 = (unsigned)(size_t)&As[0][tid * 40];
  const unsigned aoff1 = (unsigned)(size_t)&As[1][tid * 40];
  const unsigned boff0 = (unsigned)(size_t)&Bs[0][bnoff];
  const unsigned boff1 = (unsigned)(size_t)&Bs[1][bnoff];

  auto issue = [&](int k0, int b) {
    const void* g = (const void*)(aptr + k0);
    unsigned ao = b ? aoff1 : aoff0;
    async_cp16<0>(g, ao);
    async_cp16<16>(g, ao);
    async_cp16<32>(g, ao);
    async_cp16<48>(g, ao);
    if (TRANSB) {
      const void* gb = (const void*)(bptr + k0);
      unsigned bo = b ? boff1 : boff0;
      async_cp16<0>(gb, bo);
      async_cp16<16>(gb, bo);
    }
  };

  issue(0, 0);
  int ib = 0;
  for (int k0 = 0; k0 < K; k0 += 32, ib ^= 1) {
    const bool more = (k0 + 32 < K);
    if (more) issue(k0 + 32, ib ^ 1);
    if (!TRANSB) {
      // synchronous B staging (transpose scatter) into the current buffer
      const f16* bp = bptr + (long)k0 * N;
      v8h r0 = *(const v8h*)(bp);
      v8h r1 = *(const v8h*)(bp + N);
      f16* bsb = &Bs[ib][0];
#pragma unroll
      for (int j = 0; j < 8; ++j) {
        f16* d = bsb + (bc8 + j) * 40 + bk2;    // adjacent pair -> ds_store_b32
        d[0] = r0[j];
        d[1] = r1[j];
      }
    }
    if (more) __builtin_amdgcn_s_wait_asynccnt(NISS);
    else      __builtin_amdgcn_s_wait_asynccnt(0);
    __syncthreads();

    const f16* as = &As[ib][0];
    const f16* bs = &Bs[ib][0];
    v16h af0 = frag_a(as + (wave * 32)      * 40, 40, lane);
    v16h af1 = frag_a(as + (wave * 32 + 16) * 40, 40, lane);
#pragma unroll
    for (int j = 0; j < 4; ++j) {
      v16h bf = frag_b(bs, 40, j * 16, lane);
      acc[0][j] = wmma_f16(af0, bf, acc[0][j]);
      acc[1][j] = wmma_f16(af1, bf, acc[1][j]);
    }
    __syncthreads();
  }
#else
  // ================= fallback: register-staged single buffer =================
  __shared__ __align__(16) f16 As[128 * 40];
  __shared__ __align__(16) f16 Bs[64 * 40];
  f16* asrow = As + tid * 40;

  for (int k0 = 0; k0 < K; k0 += 32) {
    {
      const f16* ap = aptr + k0;
      v8h a0 = *(const v8h*)(ap);
      v8h a1 = *(const v8h*)(ap + 8);
      v8h a2 = *(const v8h*)(ap + 16);
      v8h a3 = *(const v8h*)(ap + 24);
      *(v8h*)(asrow)      = a0;
      *(v8h*)(asrow + 8)  = a1;
      *(v8h*)(asrow + 16) = a2;
      *(v8h*)(asrow + 24) = a3;
    }
    if (TRANSB) {
      v8h b0 = *(const v8h*)(bptr + k0);
      v8h b1 = *(const v8h*)(bptr + k0 + 8);
      *(v8h*)(Bs + bnoff)     = b0;
      *(v8h*)(Bs + bnoff + 8) = b1;
    } else {
      const f16* bp = bptr + (long)k0 * N;
      v8h r0 = *(const v8h*)(bp);
      v8h r1 = *(const v8h*)(bp + N);
#pragma unroll
      for (int j = 0; j < 8; ++j) {
        f16* d = Bs + (bc8 + j) * 40 + bk2;
        d[0] = r0[j];
        d[1] = r1[j];
      }
    }
    if (k0 + 32 < K) __builtin_prefetch(aptr + k0 + 32, 0, 1);
    __syncthreads();

    v16h af0 = frag_a(As + (wave * 32)      * 40, 40, lane);
    v16h af1 = frag_a(As + (wave * 32 + 16) * 40, 40, lane);
#pragma unroll
    for (int j = 0; j < 4; ++j) {
      v16h bf = frag_b(Bs, 40, j * 16, lane);
      acc[0][j] = wmma_f16(af0, bf, acc[0][j]);
      acc[1][j] = wmma_f16(af1, bf, acc[1][j]);
    }
    __syncthreads();
  }
#endif

  // ---- epilogue: vgpr r -> M = r + 8*half, N = lane&15
  const int half = lane >> 4;
  const int nnb  = n0 + (lane & 15);
#pragma unroll
  for (int i = 0; i < 2; ++i) {
#pragma unroll
    for (int j = 0; j < 4; ++j) {
      int nn = nnb + j * 16;
#pragma unroll
      for (int r = 0; r < 8; ++r) {
        int gm = m0 + wave * 32 + i * 16 + r + 8 * half;
        if (gm < M) {
          long ci = cb + (long)gm * N + nn;
          float v = acc[i][j][r] * alpha;
          if (bias) v += bias[nn];
          if (act == 1)      v = fmaxf(v, 0.0f);
          else if (act == 2) v = 1.0f / (1.0f + __expf(-v));
          if (mulsrc) v *= mulsrc[ci];
          if (gtab)   v += (float)gtab[(long)gidx[(long)bz * M + gm] * N + nn];
          if (resid)  v += resid[ci];
          if (outF) outF[ci] = v;
          if (outH) outH[ci] = (f16)v;
        }
      }
    }
  }
}

// ---------------------------------------------------------------------------
// f32 -> f16 conversion (row-major copy)
// ---------------------------------------------------------------------------
__global__ void __launch_bounds__(256)
cvt_kernel(const float* __restrict__ in, f16* __restrict__ o, long n) {
  long i = (long)blockIdx.x * 1024 + threadIdx.x;
#pragma unroll
  for (int t = 0; t < 4; ++t) {
    long j = i + (long)t * 256;
    if (j < n) o[j] = (f16)in[j];
  }
}

// f32 (rows x cols) -> f16 transposed (cols x rows), 32x32 LDS tiles
__global__ void __launch_bounds__(256)
cvtT_kernel(const float* __restrict__ in, f16* __restrict__ o, int rows, int cols) {
  __shared__ float t[32][33];
  const int bx = blockIdx.x * 32;     // col tile
  const int by = blockIdx.y * 32;     // row tile
  const int tx = threadIdx.x & 31;
  const int ty = threadIdx.x >> 5;    // 8 rows per pass
#pragma unroll
  for (int i = 0; i < 32; i += 8) {
    int r = by + ty + i, c = bx + tx;
    t[ty + i][tx] = (r < rows && c < cols) ? in[(long)r * cols + c] : 0.0f;
  }
  __syncthreads();
#pragma unroll
  for (int i = 0; i < 32; i += 8) {
    int r = bx + ty + i, c = by + tx;  // out is cols x rows
    if (r < cols && c < rows) o[(long)r * rows + c] = (f16)t[tx][ty + i];
  }
}

// ---------------------------------------------------------------------------
// LayerNorm over last dim (512), one block (256 thr) per row, f16 output
// ---------------------------------------------------------------------------
__global__ void __launch_bounds__(256)
ln_kernel(const float* __restrict__ x, const float* __restrict__ g,
          const float* __restrict__ b, f16* __restrict__ o)
{
  __shared__ float s1[256], s2[256];
  const long base = (long)blockIdx.x * D_MODEL;
  const int tid = threadIdx.x;
  float v0 = x[base + tid];
  float v1 = x[base + tid + 256];
  s1[tid] = v0 + v1;
  s2[tid] = v0 * v0 + v1 * v1;
  __syncthreads();
  for (int s = 128; s > 0; s >>= 1) {
    if (tid < s) { s1[tid] += s1[tid + s]; s2[tid] += s2[tid + s]; }
    __syncthreads();
  }
  const float mu  = s1[0] * (1.0f / D_MODEL);
  const float var = s2[0] * (1.0f / D_MODEL) - mu * mu;
  const float rs  = rsqrtf(var + 1e-5f);
  o[base + tid]       = (f16)((v0 - mu) * rs * g[tid] + b[tid]);
  o[base + tid + 256] = (f16)((v1 - mu) * rs * g[tid + 256] + b[tid + 256]);
}

// ---------------------------------------------------------------------------
// Forward attention weights: per (b,n) row over L=256
// ---------------------------------------------------------------------------
__global__ void __launch_bounds__(256)
fw_kernel(const float* __restrict__ z, const int* __restrict__ mask,
          f16* __restrict__ afw, f16* __restrict__ bfw)
{
  __shared__ float sa[256], sb[256];
  const long base = (long)blockIdx.x * LEN;
  const int tid = threadIdx.x;
  float zv = z[base + tid];
  int   m  = mask[base + tid];
  float a  = (m != 0 && zv > 0.0f) ?  zv : 0.0f;
  float bb = (m != 0 && zv < 0.0f) ? -zv : 0.0f;
  sa[tid] = a; sb[tid] = bb;
  __syncthreads();
  for (int s = 128; s > 0; s >>= 1) {
    if (tid < s) { sa[tid] += sa[tid + s]; sb[tid] += sb[tid + s]; }
    __syncthreads();
  }
  const float ra = 1.0f / (sa[0] + 1e-9f);
  const float rb = 1.0f / (sb[0] + 1e-9f);
  afw[base + tid] = (f16)(a  * ra);
  bfw[base + tid] = (f16)(bb * rb);
}

// ---------------------------------------------------------------------------
// Backward attention weights: per (b,l), reduce over N=992, transposed output
// ---------------------------------------------------------------------------
__global__ void __launch_bounds__(256)
bw_kernel(const float* __restrict__ z, const int* __restrict__ mask,
          f16* __restrict__ abw, f16* __restrict__ bbw)
{
  __shared__ float sa[256], sb[256];
  const int b = blockIdx.x >> 8;
  const int l = blockIdx.x & 255;
  const float* zb = z    + (long)b * N_EXP * LEN + l;
  const int*   mb = mask + (long)b * N_EXP * LEN + l;
  const int tid = threadIdx.x;

  float av[4], bv[4];
  float suma = 0.0f, sumb = 0.0f;
#pragma unroll
  for (int i = 0; i < 4; ++i) {
    int n = tid + i * 256;
    float a = 0.0f, bb = 0.0f;
    if (n < N_EXP) {
      float zv = zb[(long)n * LEN];
      int   m  = mb[(long)n * LEN];
      a  = (m != 0 && zv > 0.0f) ?  zv : 0.0f;
      bb = (m != 0 && zv < 0.0f) ? -zv : 0.0f;
    }
    av[i] = a; bv[i] = bb; suma += a; sumb += bb;
  }
  sa[tid] = suma; sb[tid] = sumb;
  __syncthreads();
  for (int s = 128; s > 0; s >>= 1) {
    if (tid < s) { sa[tid] += sa[tid + s]; sb[tid] += sb[tid + s]; }
    __syncthreads();
  }
  const float ra = 1.0f / (sa[0] + 1e-9f);
  const float rb = 1.0f / (sb[0] + 1e-9f);
  f16* ao = abw + ((long)b * LEN + l) * N_EXP;
  f16* bo = bbw + ((long)b * LEN + l) * N_EXP;
#pragma unroll
  for (int i = 0; i < 4; ++i) {
    int n = tid + i * 256;
    if (n < N_EXP) { ao[n] = (f16)(av[i] * ra); bo[n] = (f16)(bv[i] * rb); }
  }
}

// ---------------------------------------------------------------------------
// Residual + sel blend: out = x + sel*oa + (1-sel)*ob
// ---------------------------------------------------------------------------
__global__ void __launch_bounds__(256)
combine_kernel(const float* __restrict__ x, const float* __restrict__ sel,
               const float* __restrict__ oa, const float* __restrict__ ob,
               float* __restrict__ out, long n)
{
  long i = (long)blockIdx.x * 256 + threadIdx.x;
  if (i < n) {
    float s = sel[i];
    out[i] = x[i] + s * oa[i] + (1.0f - s) * ob[i];
  }
}

// ---------------------------------------------------------------------------
// Host launcher
// ---------------------------------------------------------------------------
extern "C" void kernel_launch(void* const* d_in, const int* in_sizes, int n_in,
                              void* d_out, int out_size, void* d_ws, size_t ws_size,
                              hipStream_t stream)
{
  (void)in_sizes; (void)n_in; (void)out_size; (void)ws_size;

  const float* x     = (const float*)d_in[0];
  const int*   nidx  = (const int*)  d_in[1];
  const int*   mask  = (const int*)  d_in[2];
  const float* ln1g  = (const float*)d_in[3];
  const float* ln1b  = (const float*)d_in[4];
  const float* ln2g  = (const float*)d_in[5];
  const float* ln2b  = (const float*)d_in[6];
  const float* q_tab = (const float*)d_in[7];
  const float* b_tab = (const float*)d_in[8];
  const float* Wk  = (const float*)d_in[9];   const float* bk   = (const float*)d_in[10];
  const float* Wa  = (const float*)d_in[11];  const float* ba   = (const float*)d_in[12];
  const float* Wa1 = (const float*)d_in[13];  const float* ba1  = (const float*)d_in[14];
  const float* Wb  = (const float*)d_in[15];  const float* bb   = (const float*)d_in[16];
  const float* Wb1 = (const float*)d_in[17];  const float* bb1  = (const float*)d_in[18];
  const float* Ws  = (const float*)d_in[19];  const float* bsel = (const float*)d_in[20];
  const float* Wf1 = (const float*)d_in[21];  const float* bf1  = (const float*)d_in[22];
  const float* Wf2 = (const float*)d_in[23];  const float* bf2  = (const float*)d_in[24];
  float* out = (float*)d_out;

  char* p = (char*)d_ws;
  auto alloc = [&](size_t elems, size_t esz) -> void* {
    void* r = (void*)p;
    p += (elems * esz + 255) & ~(size_t)255;
    return r;
  };

  f16* WkT   = (f16*)alloc((size_t)D_MODEL * D_MODEL, 2);
  f16* WaT   = (f16*)alloc((size_t)D_MODEL * D_MODEL, 2);
  f16* Wa1T  = (f16*)alloc((size_t)D_MODEL * D_MODEL, 2);
  f16* WbT   = (f16*)alloc((size_t)D_MODEL * D_MODEL, 2);
  f16* Wb1T  = (f16*)alloc((size_t)D_MODEL * D_MODEL, 2);
  f16* WsT   = (f16*)alloc((size_t)D_MODEL * D_MODEL, 2);
  f16* Wf1T  = (f16*)alloc((size_t)D_MODEL * D_FF,    2);
  f16* Wf2T  = (f16*)alloc((size_t)D_FF * D_MODEL,    2);
  f16* qth   = (f16*)alloc((size_t)N_EXP * D_MODEL,   2);
  f16* bth   = (f16*)alloc((size_t)N_EXP * D_MODEL,   2);

  f16*   x2h   = (f16*)  alloc((size_t)ROWS * D_MODEL, 2);
  f16*   xkh   = (f16*)  alloc((size_t)ROWS * D_MODEL, 2);
  float* apref = (float*)alloc((size_t)ROWS * D_MODEL, 4);
  float* bpref = (float*)alloc((size_t)ROWS * D_MODEL, 4);
  f16*   apreh = (f16*)  alloc((size_t)ROWS * D_MODEL, 2);
  f16*   bpreh = (f16*)  alloc((size_t)ROWS * D_MODEL, 2);
  f16*   aembh = (f16*)  alloc((size_t)ROWS * D_MODEL, 2);
  f16*   bembh = (f16*)  alloc((size_t)ROWS * D_MODEL, 2);
  float* self_ = (float*)alloc((size_t)ROWS * D_MODEL, 4);
  float* zf    = (float*)alloc((size_t)BS * N_EXP * LEN, 4);
  f16*   afwh  = (f16*)  alloc((size_t)BS * N_EXP * LEN, 2);
  f16*   bfwh  = (f16*)  alloc((size_t)BS * N_EXP * LEN, 2);
  f16*   clah  = (f16*)  alloc((size_t)BS * N_EXP * D_MODEL, 2);
  f16*   clbh  = (f16*)  alloc((size_t)BS * N_EXP * D_MODEL, 2);
  f16*   abwh  = (f16*)  alloc((size_t)BS * LEN * N_EXP, 2);
  f16*   bbwh  = (f16*)  alloc((size_t)BS * LEN * N_EXP, 2);
  float* outaf = (float*)alloc((size_t)ROWS * D_MODEL, 4);
  float* outbf = (float*)alloc((size_t)ROWS * D_MODEL, 4);
  f16*   x3h   = (f16*)  alloc((size_t)ROWS * D_MODEL, 2);
  f16*   ffh   = (f16*)  alloc((size_t)ROWS * D_FF,    2);

  auto cvt = [&](const float* in, f16* o, long n) {
    cvt_kernel<<<dim3((unsigned)((n + 1023) / 1024)), 256, 0, stream>>>(in, o, n);
  };
  auto cvtT = [&](const float* in, f16* o, int rows, int cols) {
    dim3 g((unsigned)((cols + 31) / 32), (unsigned)((rows + 31) / 32));
    cvtT_kernel<<<g, 256, 0, stream>>>(in, o, rows, cols);
  };
  auto gemm = [&](const f16* A, const f16* B, float* oF, f16* oH,
                  const float* bias_, const float* mul_, const float* res_,
                  const f16* gt, const int* gi, const int* ag,
                  int M, int N, int K, long sA, long sB, long sC,
                  int batch, int tB, float alpha, int act) {
    dim3 g((unsigned)((M + 127) / 128), (unsigned)(N / 64), (unsigned)batch);
    if (tB)
      gemm_wmma_kernel<true><<<g, 128, 0, stream>>>(A, B, oF, oH, bias_, mul_, res_,
                                                    gt, gi, ag, M, N, K, sA, sB, sC, alpha, act);
    else
      gemm_wmma_kernel<false><<<g, 128, 0, stream>>>(A, B, oF, oH, bias_, mul_, res_,
                                                     gt, gi, ag, M, N, K, sA, sB, sC, alpha, act);
  };

  // 0) weights -> transposed f16 (N x K layout for the async TRANSB path)
  cvtT(Wk,  WkT,  D_MODEL, D_MODEL);
  cvtT(Wa,  WaT,  D_MODEL, D_MODEL);
  cvtT(Wa1, Wa1T, D_MODEL, D_MODEL);
  cvtT(Wb,  WbT,  D_MODEL, D_MODEL);
  cvtT(Wb1, Wb1T, D_MODEL, D_MODEL);
  cvtT(Ws,  WsT,  D_MODEL, D_MODEL);
  cvtT(Wf1, Wf1T, D_MODEL, D_FF);
  cvtT(Wf2, Wf2T, D_FF, D_MODEL);
  cvt(q_tab, qth, (long)N_EXP * D_MODEL);
  cvt(b_tab, bth, (long)N_EXP * D_MODEL);

  // 1) x2 = LN1(x)
  ln_kernel<<<ROWS, 256, 0, stream>>>(x, ln1g, ln1b, x2h);

  // 2) projections of x2 (TRANSB: weight rows are output features)
  gemm(x2h, WkT, nullptr, xkh, bk, nullptr, nullptr, nullptr, nullptr, nullptr,
       ROWS, D_MODEL, D_MODEL, 0, 0, 0, 1, 1, 1.0f, 0);
  gemm(x2h, WaT, apref, apreh, ba, nullptr, nullptr, nullptr, nullptr, nullptr,
       ROWS, D_MODEL, D_MODEL, 0, 0, 0, 1, 1, 1.0f, 0);
  gemm(x2h, WbT, bpref, bpreh, bb, nullptr, nullptr, nullptr, nullptr, nullptr,
       ROWS, D_MODEL, D_MODEL, 0, 0, 0, 1, 1, 1.0f, 0);
  gemm(x2h, WsT, self_, nullptr, bsel, nullptr, nullptr, nullptr, nullptr, nullptr,
       ROWS, D_MODEL, D_MODEL, 0, 0, 0, 1, 1, 1.0f, 2);          // sel = sigmoid

  // 3) gated embeddings: a_emb = a_pre * sigmoid(a_pre @ Wa1 + ba1)
  gemm(apreh, Wa1T, nullptr, aembh, ba1, apref, nullptr, nullptr, nullptr, nullptr,
       ROWS, D_MODEL, D_MODEL, 0, 0, 0, 1, 1, 1.0f, 2);
  gemm(bpreh, Wb1T, nullptr, bembh, bb1, bpref, nullptr, nullptr, nullptr, nullptr,
       ROWS, D_MODEL, D_MODEL, 0, 0, 0, 1, 1, 1.0f, 2);

  // 4) z[b] = q_tab[n_indexes[b]] @ x_key[b]^T / sqrt(D)
  gemm(qth, xkh, zf, nullptr, nullptr, nullptr, nullptr, nullptr, nullptr, nidx,
       N_EXP, LEN, D_MODEL, 0, (long)LEN * D_MODEL, (long)N_EXP * LEN,
       BS, 1, 0.044194173824159216f, 0);

  // 5) forward masked-relu row-normalize
  fw_kernel<<<BS * N_EXP, 256, 0, stream>>>(zf, mask, afwh, bfwh);

  // 6) class_a[b] = a_fw[b] @ a_emb[b] + b_tab[n_indexes[b]]
  gemm(afwh, aembh, nullptr, clah, nullptr, nullptr, nullptr, bth, nidx, nullptr,
       N_EXP, D_MODEL, LEN, (long)N_EXP * LEN, (long)LEN * D_MODEL, (long)N_EXP * D_MODEL,
       BS, 0, 1.0f, 0);
  gemm(bfwh, bembh, nullptr, clbh, nullptr, nullptr, nullptr, bth, nidx, nullptr,
       N_EXP, D_MODEL, LEN, (long)N_EXP * LEN, (long)LEN * D_MODEL, (long)N_EXP * D_MODEL,
       BS, 0, 1.0f, 0);

  // 7) backward masked-relu column-normalize (transposed output)
  bw_kernel<<<BS * LEN, 256, 0, stream>>>(zf, mask, abwh, bbwh);

  // 8) out_a[b] = a_bw[b] @ class_a[b]   (K = 992 = 31*32)
  gemm(abwh, clah, outaf, nullptr, nullptr, nullptr, nullptr, nullptr, nullptr, nullptr,
       LEN, D_MODEL, N_EXP, (long)LEN * N_EXP, (long)N_EXP * D_MODEL, (long)LEN * D_MODEL,
       BS, 0, 1.0f, 0);
  gemm(bbwh, clbh, outbf, nullptr, nullptr, nullptr, nullptr, nullptr, nullptr, nullptr,
       LEN, D_MODEL, N_EXP, (long)LEN * N_EXP, (long)N_EXP * D_MODEL, (long)LEN * D_MODEL,
       BS, 0, 1.0f, 0);

  // 9) x_mid = x + sel*out_a + (1-sel)*out_b   -> d_out
  combine_kernel<<<(unsigned)(((long)ROWS * D_MODEL + 255) / 256), 256, 0, stream>>>(
      x, self_, outaf, outbf, out, (long)ROWS * D_MODEL);

  // 10) FF: x3 = LN2(x_mid); out = x_mid + relu(x3@Wf1+bf1)@Wf2 + bf2
  ln_kernel<<<ROWS, 256, 0, stream>>>(out, ln2g, ln2b, x3h);
  gemm(x3h, Wf1T, nullptr, ffh, bf1, nullptr, nullptr, nullptr, nullptr, nullptr,
       ROWS, D_FF, D_MODEL, 0, 0, 0, 1, 1, 1.0f, 1);             // relu
  gemm(ffh, Wf2T, out, nullptr, bf2, nullptr, out, nullptr, nullptr, nullptr,
       ROWS, D_MODEL, D_FF, 0, 0, 0, 1, 1, 1.0f, 0);             // + residual
}
